// PARSeq_46153718563326
// MI455X (gfx1250) — matrix operand order repaired and checked by reading, not compiled, not run on hardware
//
#include <hip/hip_runtime.h>
#include <hip/hip_bf16.h>
#include <math.h>

// ---------------- model constants ----------------
#define DMOD   384
#define MLPD   1536
#define NTOK   128
#define BSZ    32
#define EHEADS 6      // enc heads, hd = 64
#define DHEADS 12     // dec heads, hd = 32
#define NS     26     // max_length + 1
#define VOCAB  95     // NUM_TOKENS - 2
#define NTOKENS 97
#define KNN    8

typedef __attribute__((ext_vector_type(16))) _Float16 v16h;
typedef __attribute__((ext_vector_type(8)))  float    v8f;

// ---------------- input leaf indices -------------------------------------
// Assumption: the harness flattens setup_inputs() like jax.tree_util
// (dict keys sorted, depth-first). If not, only this enum needs remapping.
enum {
  IN_BOS = 0, IN_EOS, IN_EVIMG, IN_MAXLEN, IN_PAD,
  IN_DEC_CA_BK, IN_DEC_CA_BO, IN_DEC_CA_BQ, IN_DEC_CA_BV,
  IN_DEC_CA_WK, IN_DEC_CA_WO, IN_DEC_CA_WQ, IN_DEC_CA_WV,
  IN_DEC_L1B, IN_DEC_L1W, IN_DEC_L2B, IN_DEC_L2W,
  IN_DEC_N1B, IN_DEC_N1G, IN_DEC_N2B, IN_DEC_N2G,
  IN_DEC_NCB, IN_DEC_NCG, IN_DEC_NFB, IN_DEC_NFG, IN_DEC_NQB, IN_DEC_NQG,
  IN_DEC_SA_BK, IN_DEC_SA_BO, IN_DEC_SA_BQ, IN_DEC_SA_BV,
  IN_DEC_SA_WK, IN_DEC_SA_WO, IN_DEC_SA_WQ, IN_DEC_SA_WV,
  IN_ENC_FC1B, IN_ENC_FC1W, IN_ENC_FC2B, IN_ENC_FC2W,
  IN_ENC_LN1B, IN_ENC_LN1G, IN_ENC_LN2B, IN_ENC_LN2G,
  IN_ENC_PROJB, IN_ENC_PROJW, IN_ENC_QKVB, IN_ENC_QKVW,
  IN_ENC_NORMB, IN_ENC_NORMG, IN_ENC_POS,
  IN_HEADB, IN_HEADW,
  IN_HG_B1, IN_HG_B2, IN_HG_W1, IN_HG_W2,
  IN_PATCHB, IN_PATCHW, IN_POSQ, IN_TEXTEMB, IN_RGBIMG
};

// ---------------- device helpers -----------------------------------------
static __device__ __forceinline__ float gelu_f(float x) {
  float x3 = x * x * x;
  return 0.5f * x * (1.0f + tanhf(0.7978845608f * (x + 0.044715f * x3)));
}

// A fragment (16x32 f16, row-major fp32 source), contiguous k order, same as fragB:
// lane<16 holds K[k0..k0+15], lane>=16 holds K[k0+16..k0+31].
static __device__ __forceinline__ v16h fragA_f32(const float* A, int ld,
                                                 int row0, int k0, int rmax,
                                                 int lane) {
  int r = row0 + (lane & 15);
  if (r > rmax) r = rmax;
  int kb = k0 + ((lane & 16) ? 16 : 0);
  const float* p = A + (long)r * ld;
  v16h f;
#pragma unroll
  for (int i = 0; i < 8; ++i) f[i] = (_Float16)p[kb + i];
#pragma unroll
  for (int i = 0; i < 8; ++i) f[8 + i] = (_Float16)p[kb + 8 + i];
  return f;
}

// B fragment (32x16 f16): lane<16 holds K[k0..k0+15] of column (lane&15),
// lane>=16 holds K[k0+16..k0+31].  bTrans!=0 => B stored N x K row-major.
static __device__ __forceinline__ v16h fragB_f32(const float* B, int ld,
                                                 int k0, int col0, int cmax,
                                                 int lane, int bTrans) {
  int c = col0 + (lane & 15);
  if (c > cmax) c = cmax;
  int kb = k0 + ((lane & 16) ? 16 : 0);
  v16h f;
  if (!bTrans) {
#pragma unroll
    for (int i = 0; i < 16; ++i) f[i] = (_Float16)B[(long)(kb + i) * ld + c];
  } else {
#pragma unroll
    for (int i = 0; i < 16; ++i) f[i] = (_Float16)B[(long)c * ld + (kb + i)];
  }
  return f;
}

static __device__ __forceinline__ v8f wmma_f16(v16h a, v16h b, v8f c) {
  return __builtin_amdgcn_wmma_f32_16x16x32_f16(false, a, false, b, (short)0, c,
                                                false, false);
}

// ---------------- generic WMMA GEMM ---------------------------------------
// C = act(A @ B + bias) + Res.  One wave per 16x64 output macro-tile:
// the A fragment is reused across 4 WMMAs per k-step (4x arithmetic
// intensity vs a single 16x16 tile; the 4 accumulators are independent so
// the WMMAs pipeline without D->A/B hazard NOPs).  K % 32 == 0.
// blockIdx.y = batch; per-batch element strides sA/sB/sC/sRes.
__global__ void gemm_kernel(const float* __restrict__ A,
                            const float* __restrict__ B,
                            const float* __restrict__ bias,
                            const float* __restrict__ Res,
                            float* __restrict__ C, int M, int N, int K, int lda,
                            int ldb, int ldc, long sA, long sB, long sC,
                            long sRes, int bTrans, int act) {
  int bat = blockIdx.y;
  A += (long)bat * sA;
  B += (long)bat * sB;
  C += (long)bat * sC;
  if (Res) Res += (long)bat * sRes;

  int wave = threadIdx.x >> 5;
  int lane = threadIdx.x & 31;
  int tilesNg = (N + 63) >> 6;  // groups of 4 col-tiles (64 cols)
  int tilesM = (M + 15) >> 4;
  int tile = blockIdx.x * (blockDim.x >> 5) + wave;
  if (tile >= tilesM * tilesNg) return;  // wave-uniform exit
  int tm = tile / tilesNg, tg = tile % tilesNg;
  int row0 = tm << 4, col0 = tg << 6;

  v8f acc0 = {0.f, 0.f, 0.f, 0.f, 0.f, 0.f, 0.f, 0.f};
  v8f acc1 = acc0, acc2 = acc0, acc3 = acc0;
  for (int k0 = 0; k0 < K; k0 += 32) {
    if (!bTrans && (k0 + 32 < K))
      __builtin_prefetch(B + (long)(k0 + 32) * ldb + col0, 0, 0);
    v16h a = fragA_f32(A, lda, row0, k0, M - 1, lane);
    v16h b0 = fragB_f32(B, ldb, k0, col0, N - 1, lane, bTrans);
    v16h b1 = fragB_f32(B, ldb, k0, col0 + 16, N - 1, lane, bTrans);
    v16h b2 = fragB_f32(B, ldb, k0, col0 + 32, N - 1, lane, bTrans);
    v16h b3 = fragB_f32(B, ldb, k0, col0 + 48, N - 1, lane, bTrans);
    acc0 = wmma_f16(a, b0, acc0);
    acc1 = wmma_f16(a, b1, acc1);
    acc2 = wmma_f16(a, b2, acc2);
    acc3 = wmma_f16(a, b3, acc3);
  }
  int rbase = row0 + ((lane & 16) ? 8 : 0);
#pragma unroll
  for (int t = 0; t < 4; ++t) {
    v8f acc = (t == 0) ? acc0 : (t == 1) ? acc1 : (t == 2) ? acc2 : acc3;
    int col = col0 + t * 16 + (lane & 15);
    float bcol = (bias != nullptr && col < N) ? bias[col] : 0.f;
#pragma unroll
    for (int r = 0; r < 8; ++r) {
      int row = rbase + r;
      if (row < M && col < N) {
        float v = acc[r] + bcol;
        if (act == 1) v = gelu_f(v);
        else if (act == 2) v = fmaxf(v, 0.f);
        if (Res) v += Res[(long)row * ldc + col];
        C[(long)row * ldc + col] = v;
      }
    }
  }
}

// ---------------- LayerNorm ------------------------------------------------
__global__ void ln_kernel(const float* __restrict__ x, const float* __restrict__ g,
                          const float* __restrict__ b, float* __restrict__ y,
                          int rows, int Dm) {
  int row = blockIdx.x;
  if (row >= rows) return;
  const float* xr = x + (long)row * Dm;
  float* yr = y + (long)row * Dm;
  __shared__ float red[128];
  int tid = threadIdx.x;
  float s = 0.f;
  for (int i = tid; i < Dm; i += 128) s += xr[i];
  red[tid] = s;
  __syncthreads();
  for (int o = 64; o > 0; o >>= 1) {
    if (tid < o) red[tid] += red[tid + o];
    __syncthreads();
  }
  float mean = red[0] / Dm;
  __syncthreads();
  float v = 0.f;
  for (int i = tid; i < Dm; i += 128) {
    float d = xr[i] - mean;
    v += d * d;
  }
  red[tid] = v;
  __syncthreads();
  for (int o = 64; o > 0; o >>= 1) {
    if (tid < o) red[tid] += red[tid + o];
    __syncthreads();
  }
  float rstd = rsqrtf(red[0] / Dm + 1e-5f);
  for (int i = tid; i < Dm; i += 128) yr[i] = (xr[i] - mean) * rstd * g[i] + b[i];
}

// ---------------- patchify -------------------------------------------------
__global__ void patchify_kernel(const float* __restrict__ img,
                                float* __restrict__ tok96) {
  long i = (long)blockIdx.x * 256 + threadIdx.x;
  long total = (long)BSZ * NTOK * 96;
  if (i >= total) return;
  int f = (int)(i % 96);
  long t2 = i / 96;
  int t = (int)(t2 % NTOK);
  int b = (int)(t2 / NTOK);
  int c = f / 32, rem = f % 32, ph = rem / 8, pw = rem % 8;
  int gh = t / 16, gw = t % 16;  // GH=8, GW=16
  tok96[i] = img[(((long)b * 3 + c) * 32 + gh * 4 + ph) * 128 + gw * 8 + pw];
}

// x = tok + pos (+ hg)
__global__ void add_pos_kernel(const float* __restrict__ tok,
                               const float* __restrict__ pos,
                               const float* __restrict__ hg,
                               float* __restrict__ x, long total) {
  long i = (long)blockIdx.x * 256 + threadIdx.x;
  if (i >= total) return;
  float v = tok[i] + pos[i % ((long)NTOK * DMOD)];
  if (hg) v += hg[i];
  x[i] = v;
}

__global__ void concat_kernel(const float* __restrict__ a,
                              const float* __restrict__ b,
                              float* __restrict__ c, long rows) {
  long i = (long)blockIdx.x * 256 + threadIdx.x;
  if (i >= rows * 768) return;
  long r = i / 768;
  int f = (int)(i % 768);
  c[i] = (f < DMOD) ? a[r * DMOD + f] : b[r * DMOD + (f - DMOD)];
}

// ---------------- encoder attention (flash-in-LDS, WMMA) -------------------
// one block per (batch, head); L=128, hd=64
__global__ void attn_enc_kernel(const float* __restrict__ qkv,
                                float* __restrict__ out) {
  __shared__ _Float16 KT[64][128];   // K^T : [hd][L]
  __shared__ _Float16 Vs[128][64];   // V   : [L][hd]
  __shared__ float S[128][132];      // scores (padded)

  int bh = blockIdx.x;
  int h = bh % EHEADS;
  int b = bh / EHEADS;
  int tid = threadIdx.x;
  int lane = tid & 31;
  int wave = tid >> 5;

  const float* base = qkv + (long)b * NTOK * 3 * DMOD;
  for (int idx = tid; idx < 128 * 64; idx += 256) {
    int n = idx >> 6, d = idx & 63;
    KT[d][n] = (_Float16)base[(long)n * 3 * DMOD + DMOD + h * 64 + d];
    Vs[n][d] = (_Float16)base[(long)n * 3 * DMOD + 2 * DMOD + h * 64 + d];
  }
  __syncthreads();

  // S = (Q @ K^T) / 8 ; wave w owns row-tile w, sweeps 8 col-tiles
  {
    int tm = wave;
    for (int tn = 0; tn < 8; ++tn) {
      v8f acc = {0.f, 0.f, 0.f, 0.f, 0.f, 0.f, 0.f, 0.f};
      for (int k0 = 0; k0 < 64; k0 += 32) {
        int r = tm * 16 + (lane & 15);
        int kb = k0 + ((lane & 16) ? 16 : 0);
        const float* qp = base + (long)r * 3 * DMOD + h * 64;
        v16h af;
#pragma unroll
        for (int i = 0; i < 8; ++i) af[i] = (_Float16)qp[kb + i];
#pragma unroll
        for (int i = 0; i < 8; ++i) af[8 + i] = (_Float16)qp[kb + 8 + i];
        int c = tn * 16 + (lane & 15);
        int kb2 = k0 + ((lane & 16) ? 16 : 0);
        v16h bf;
#pragma unroll
        for (int i = 0; i < 16; ++i) bf[i] = KT[kb2 + i][c];
        acc = wmma_f16(af, bf, acc);
      }
      int cc = tn * 16 + (lane & 15);
      int rb = tm * 16 + ((lane & 16) ? 8 : 0);
#pragma unroll
      for (int r = 0; r < 8; ++r) S[rb + r][cc] = acc[r] * 0.125f;
    }
  }
  __syncthreads();

  // softmax along rows
  if (tid < 128) {
    float m = -3e38f;
    for (int j = 0; j < 128; ++j) m = fmaxf(m, S[tid][j]);
    float sum = 0.f;
    for (int j = 0; j < 128; ++j) {
      float e = __expf(S[tid][j] - m);
      S[tid][j] = e;
      sum += e;
    }
    float inv = 1.f / sum;
    for (int j = 0; j < 128; ++j) S[tid][j] *= inv;
  }
  __syncthreads();

  // O = P @ V ; wave w owns row-tile w, 4 col-tiles of hd=64
  {
    int tm = wave;
    for (int tn = 0; tn < 4; ++tn) {
      v8f acc = {0.f, 0.f, 0.f, 0.f, 0.f, 0.f, 0.f, 0.f};
      for (int k0 = 0; k0 < 128; k0 += 32) {
        int r = tm * 16 + (lane & 15);
        int kb = k0 + ((lane & 16) ? 16 : 0);
        v16h af;
#pragma unroll
        for (int i = 0; i < 8; ++i) af[i] = (_Float16)S[r][kb + i];
#pragma unroll
        for (int i = 0; i < 8; ++i) af[8 + i] = (_Float16)S[r][kb + 8 + i];
        int c = tn * 16 + (lane & 15);
        int kb2 = k0 + ((lane & 16) ? 16 : 0);
        v16h bf;
#pragma unroll
        for (int i = 0; i < 16; ++i) bf[i] = Vs[kb2 + i][c];
        acc = wmma_f16(af, bf, acc);
      }
      int cc = tn * 16 + (lane & 15);
      int rb = tm * 16 + ((lane & 16) ? 8 : 0);
#pragma unroll
      for (int r = 0; r < 8; ++r)
        out[((long)b * NTOK + rb + r) * DMOD + h * 64 + cc] = acc[r];
    }
  }
}

// ---------------- kNN hypergraph G ----------------------------------------
// one block per batch; rgbe [B,128,768] -> G [B,128,128]
__global__ void build_g_kernel(const float* __restrict__ rgbe,
                               float* __restrict__ G) {
  __shared__ float d2[128][129];
  __shared__ int idxs[128][KNN];
  __shared__ unsigned Hb[128][4];
  __shared__ float dvis[128];
  int b = blockIdx.x;
  int tid = threadIdx.x;
  const float* x = rgbe + (long)b * NTOK * 768;

  for (int p = tid; p < 128 * 128; p += 256) {
    int n = p >> 7, m = p & 127;
    const float* xn = x + (long)n * 768;
    const float* xm = x + (long)m * 768;
    float s = 0.f;
    for (int f = 0; f < 768; ++f) {
      float d = xn[f] - xm[f];
      s += d * d;
    }
    d2[n][m] = s;
  }
  if (tid < 128) {
    Hb[tid][0] = 0u; Hb[tid][1] = 0u; Hb[tid][2] = 0u; Hb[tid][3] = 0u;
  }
  __syncthreads();

  if (tid < 128) {  // 8-NN selection per hyperedge (row)
    int e = tid;
    for (int s8 = 0; s8 < KNN; ++s8) {
      float best = 3e38f;
      int bi = 0;
      for (int m = 0; m < 128; ++m) {
        float dv = d2[e][m];
        if (dv < best) { best = dv; bi = m; }
      }
      idxs[e][s8] = bi;
      d2[e][bi] = 3e38f;
    }
  }
  __syncthreads();
  if (tid < 128) {
    int e = tid;
    for (int s8 = 0; s8 < KNN; ++s8) {
      int n = idxs[e][s8];
      atomicOr(&Hb[n][e >> 5], 1u << (e & 31));
    }
  }
  __syncthreads();
  if (tid < 128) {
    int cnt = 0;
    for (int w = 0; w < 4; ++w) cnt += __popc(Hb[tid][w]);
    dvis[tid] = rsqrtf(fmaxf((float)cnt, 1e-12f));
  }
  __syncthreads();
  for (int p = tid; p < 128 * 128; p += 256) {
    int n = p >> 7, m = p & 127;
    int cnt = 0;
    for (int w = 0; w < 4; ++w) cnt += __popc(Hb[n][w] & Hb[m][w]);
    G[(long)b * 128 * 128 + p] = dvis[n] * dvis[m] * ((float)cnt * (1.0f / KNN));
  }
}

// ---------------- decoder attention (small, VALU) --------------------------
// grid.x = B * H * Lq, block 128.  cloze: mask key t == r+1 (refinement).
__global__ void dec_attn_kernel(const float* __restrict__ q,
                                const float* __restrict__ k,
                                const float* __restrict__ v,
                                float* __restrict__ out, int B, int Lq, int Lk,
                                int hd, int Dm, const int* __restrict__ padmask,
                                int cloze) {
  __shared__ float s[128];
  __shared__ float red[128];
  int idx = blockIdx.x;
  int r = idx % Lq; idx /= Lq;
  int H = Dm / hd;
  int h = idx % H;
  int b = idx / H;
  int tid = threadIdx.x;

  const float* qp = q + ((long)b * Lq + r) * Dm + h * hd;
  float scale = rsqrtf((float)hd);
  for (int t = tid; t < Lk; t += 128) {
    const float* kp = k + ((long)b * Lk + t) * Dm + h * hd;
    float d = 0.f;
    for (int e = 0; e < hd; ++e) d += qp[e] * kp[e];
    d *= scale;
    if (cloze && t == r + 1) d = -1e9f;
    if (padmask && padmask[b * Lk + t]) d = -1e9f;
    s[t] = d;
  }
  __syncthreads();
  float m = -3e38f;
  for (int t = tid; t < Lk; t += 128) m = fmaxf(m, s[t]);
  red[tid] = m;
  __syncthreads();
  for (int o = 64; o > 0; o >>= 1) {
    if (tid < o) red[tid] = fmaxf(red[tid], red[tid + o]);
    __syncthreads();
  }
  m = red[0];
  __syncthreads();
  float sum = 0.f;
  for (int t = tid; t < Lk; t += 128) {
    float e = __expf(s[t] - m);
    s[t] = e;
    sum += e;
  }
  red[tid] = sum;
  __syncthreads();
  for (int o = 64; o > 0; o >>= 1) {
    if (tid < o) red[tid] += red[tid + o];
    __syncthreads();
  }
  float inv = 1.f / red[0];
  __syncthreads();
  for (int d0 = tid; d0 < hd; d0 += 128) {
    float o2 = 0.f;
    for (int t = 0; t < Lk; ++t)
      o2 += s[t] * v[((long)b * Lk + t) * Dm + h * hd + d0];
    out[((long)b * Lq + r) * Dm + h * hd + d0] = o2 * inv;
  }
}

// ---------------- decoder misc kernels -------------------------------------
__global__ void init_tgt_kernel(int* tgt, const int* bos, const int* pad) {
  int i = blockIdx.x * 256 + threadIdx.x;
  if (i >= BSZ * NS) return;
  tgt[i] = (i % NS == 0) ? bos[0] : pad[0];
}

__global__ void content_kernel(const int* __restrict__ tgt,
                               const float* __restrict__ emb,
                               const float* __restrict__ posq,
                               float* __restrict__ content, int j) {
  long i = (long)blockIdx.x * 256 + threadIdx.x;
  long total = (long)BSZ * j * DMOD;
  if (i >= total) return;
  int d = (int)(i % DMOD);
  int t = (int)((i / DMOD) % j);
  int b = (int)(i / ((long)DMOD * j));
  float val = sqrtf((float)DMOD) * emb[(long)tgt[b * NS + t] * DMOD + d];
  if (t > 0) val += posq[(long)(t - 1) * DMOD + d];
  content[i] = val;
}

__global__ void posq_bcast_kernel(const float* __restrict__ posq,
                                  float* __restrict__ qb, int n, int i0) {
  long i = (long)blockIdx.x * 256 + threadIdx.x;
  if (i >= (long)BSZ * n * DMOD) return;
  int d = (int)(i % DMOD);
  int t = (int)((i / DMOD) % n);
  qb[i] = posq[(long)(i0 + t) * DMOD + d];
}

__global__ void argmax_step_kernel(const float* __restrict__ logits, int* tgt,
                                   int iRow, int jcol) {
  int b = threadIdx.x;
  if (b >= BSZ) return;
  const float* p = logits + (long)b * NS * VOCAB + (long)iRow * VOCAB;
  float best = p[0];
  int bi = 0;
  for (int c = 1; c < VOCAB; ++c)
    if (p[c] > best) { best = p[c]; bi = c; }
  tgt[b * NS + jcol] = bi;
}

__global__ void retgt_kernel(const float* __restrict__ logits, int* tgt,
                             const int* bos) {
  int b = threadIdx.x;
  if (b >= BSZ) return;
  tgt[b * NS + 0] = bos[0];
  for (int t = 1; t < NS; ++t) {
    const float* p = logits + (long)b * NS * VOCAB + (long)(t - 1) * VOCAB;
    float best = p[0];
    int bi = 0;
    for (int c = 1; c < VOCAB; ++c)
      if (p[c] > best) { best = p[c]; bi = c; }
    tgt[b * NS + t] = bi;
  }
}

__global__ void padmask_kernel(const int* __restrict__ tgt, const int* eos,
                               int* __restrict__ mask) {
  int b = threadIdx.x;
  if (b >= BSZ) return;
  int acc = 0;
  for (int t = 0; t < NS; ++t) {
    if (tgt[b * NS + t] == eos[0]) acc = 1;
    mask[b * NS + t] = acc;
  }
}

// ---------------- host orchestration ---------------------------------------
static inline void gemm(hipStream_t s, const float* A, const float* B,
                        const float* bias, const float* Res, float* C, int M,
                        int N, int K, int lda, int ldb, int ldc, int batches = 1,
                        long sA = 0, long sB = 0, long sC = 0, long sRes = 0,
                        int bTrans = 0, int act = 0) {
  int tiles = ((M + 15) / 16) * ((N + 63) / 64);
  dim3 g((tiles + 7) / 8, batches, 1);
  gemm_kernel<<<g, 256, 0, s>>>(A, B, bias, Res, C, M, N, K, lda, ldb, ldc, sA,
                                sB, sC, sRes, bTrans, act);
}

struct EncW {
  const float *pos, *ln1g, *ln1b, *qkvw, *qkvb, *projw, *projb;
  const float *ln2g, *ln2b, *fc1w, *fc1b, *fc2w, *fc2b, *ng, *nb;
};

static void run_encoder(hipStream_t s, const EncW& w, const float* tok,
                        const float* hg, float* outbuf, float* x, float* h,
                        float* qkv, float* attn, float* mlph) {
  const int rows = BSZ * NTOK;
  long total = (long)rows * DMOD;
  add_pos_kernel<<<(int)((total + 255) / 256), 256, 0, s>>>(tok, w.pos, hg, x,
                                                            total);
  for (int l = 0; l < 12; ++l) {
    ln_kernel<<<rows, 128, 0, s>>>(x, w.ln1g + (long)l * DMOD,
                                   w.ln1b + (long)l * DMOD, h, rows, DMOD);
    gemm(s, h, w.qkvw + (long)l * DMOD * 3 * DMOD, w.qkvb + (long)l * 3 * DMOD,
         nullptr, qkv, rows, 3 * DMOD, DMOD, DMOD, 3 * DMOD, 3 * DMOD);
    attn_enc_kernel<<<BSZ * EHEADS, 256, 0, s>>>(qkv, attn);
    gemm(s, attn, w.projw + (long)l * DMOD * DMOD, w.projb + (long)l * DMOD, x,
         x, rows, DMOD, DMOD, DMOD, DMOD, DMOD);
    ln_kernel<<<rows, 128, 0, s>>>(x, w.ln2g + (long)l * DMOD,
                                   w.ln2b + (long)l * DMOD, h, rows, DMOD);
    gemm(s, h, w.fc1w + (long)l * DMOD * MLPD, w.fc1b + (long)l * MLPD, nullptr,
         mlph, rows, MLPD, DMOD, DMOD, MLPD, MLPD, 1, 0, 0, 0, 0, 0, /*gelu*/ 1);
    gemm(s, mlph, w.fc2w + (long)l * MLPD * DMOD, w.fc2b + (long)l * DMOD, x, x,
         rows, DMOD, MLPD, MLPD, DMOD, DMOD);
  }
  ln_kernel<<<rows, 128, 0, s>>>(x, w.ng, w.nb, outbuf, rows, DMOD);
}

extern "C" void kernel_launch(void* const* d_in, const int* in_sizes, int n_in,
                              void* d_out, int out_size, void* d_ws,
                              size_t ws_size, hipStream_t stream) {
  (void)in_sizes; (void)n_in; (void)out_size; (void)ws_size;
  hipStream_t s = stream;
  auto F = [&](int i) { return (const float*)d_in[i]; };
  auto I = [&](int i) { return (const int*)d_in[i]; };
  float* dout = (float*)d_out;  // logits [32, 26, 95]

  // workspace carve-out
  char* wp = (char*)d_ws;
  auto alloc = [&](long elems) {
    float* p = (float*)wp;
    wp += elems * sizeof(float);
    return p;
  };
  const long TD = (long)BSZ * NTOK * DMOD;
  float* tok96 = alloc((long)BSZ * NTOK * 96);
  float* ev_tok = alloc(TD);
  float* rg_tok = alloc(TD);
  float* xbuf = alloc(TD);
  float* hbuf = alloc(TD);
  float* qkvb = alloc((long)BSZ * NTOK * 3 * DMOD);
  float* attb = alloc(TD);
  float* mlpb = alloc((long)BSZ * NTOK * MLPD);
  float* ev_mem = alloc(TD);
  float* rgbe = alloc((long)BSZ * NTOK * 768);
  float* Gm = alloc((long)BSZ * NTOK * NTOK);
  float* y1 = alloc(TD);
  float* y2 = alloc(TD);
  float* hgb = alloc(TD);
  float* memry = alloc(TD);
  float* mem_k = alloc(TD);
  float* mem_v = alloc(TD);
  const long SD = (long)BSZ * NS * DMOD;
  float* content = alloc(SD);
  float* cn = alloc(SD);
  float* qb = alloc(SD);
  float* qn = alloc(SD);
  float* qv = alloc(SD);
  float* kb = alloc(SD);
  float* vb = alloc(SD);
  float* att = alloc(SD);
  float* xdec = alloc(SD);
  float* hdec = alloc(SD);
  float* mlpd = alloc((long)BSZ * NS * MLPD);
  int* tgt = (int*)alloc(BSZ * NS);
  int* pmask = (int*)alloc(BSZ * NS);

  EncW ew;
  ew.pos = F(IN_ENC_POS);
  ew.ln1g = F(IN_ENC_LN1G); ew.ln1b = F(IN_ENC_LN1B);
  ew.qkvw = F(IN_ENC_QKVW); ew.qkvb = F(IN_ENC_QKVB);
  ew.projw = F(IN_ENC_PROJW); ew.projb = F(IN_ENC_PROJB);
  ew.ln2g = F(IN_ENC_LN2G); ew.ln2b = F(IN_ENC_LN2B);
  ew.fc1w = F(IN_ENC_FC1W); ew.fc1b = F(IN_ENC_FC1B);
  ew.fc2w = F(IN_ENC_FC2W); ew.fc2b = F(IN_ENC_FC2B);
  ew.ng = F(IN_ENC_NORMG); ew.nb = F(IN_ENC_NORMB);

  const int rows = BSZ * NTOK;
  long ptot = (long)rows * 96;

  // patchify + patch-embed (W is [384, 96] row-major -> bTrans)
  patchify_kernel<<<(int)((ptot + 255) / 256), 256, 0, s>>>(F(IN_EVIMG), tok96);
  gemm(s, tok96, F(IN_PATCHW), F(IN_PATCHB), nullptr, ev_tok, rows, DMOD, 96,
       96, 96, DMOD, 1, 0, 0, 0, 0, /*bTrans*/ 1);
  patchify_kernel<<<(int)((ptot + 255) / 256), 256, 0, s>>>(F(IN_RGBIMG), tok96);
  gemm(s, tok96, F(IN_PATCHW), F(IN_PATCHB), nullptr, rg_tok, rows, DMOD, 96,
       96, 96, DMOD, 1, 0, 0, 0, 0, 1);

  // encoder pass 1 (event stream)
  run_encoder(s, ew, ev_tok, nullptr, ev_mem, xbuf, hbuf, qkvb, attb, mlpb);

  // hypergraph fusion
  concat_kernel<<<(int)(((long)rows * 768 + 255) / 256), 256, 0, s>>>(
      ev_mem, rg_tok, rgbe, rows);
  build_g_kernel<<<BSZ, 256, 0, s>>>(rgbe, Gm);
  gemm(s, rgbe, F(IN_HG_W1), nullptr, nullptr, y1, rows, DMOD, 768, 768, DMOD,
       DMOD);
  gemm(s, Gm, y1, F(IN_HG_B1), nullptr, y2, NTOK, DMOD, NTOK, NTOK, DMOD, DMOD,
       BSZ, (long)NTOK * NTOK, (long)NTOK * DMOD, (long)NTOK * DMOD, 0, 0,
       /*relu*/ 2);
  gemm(s, y2, F(IN_HG_W2), nullptr, nullptr, y1, rows, DMOD, DMOD, DMOD, DMOD,
       DMOD);
  gemm(s, Gm, y1, F(IN_HG_B2), nullptr, hgb, NTOK, DMOD, NTOK, NTOK, DMOD,
       DMOD, BSZ, (long)NTOK * NTOK, (long)NTOK * DMOD, (long)NTOK * DMOD);

  // encoder pass 2 (rgb + hypergraph injection) -> memory
  run_encoder(s, ew, rg_tok, hgb, memry, xbuf, hbuf, qkvb, attb, mlpb);

  // precompute cross-attention K/V (memory is fixed for all decode steps)
  gemm(s, memry, F(IN_DEC_CA_WK), F(IN_DEC_CA_BK), nullptr, mem_k, rows, DMOD,
       DMOD, DMOD, DMOD, DMOD);
  gemm(s, memry, F(IN_DEC_CA_WV), F(IN_DEC_CA_BV), nullptr, mem_v, rows, DMOD,
       DMOD, DMOD, DMOD, DMOD);

  // ---------------- autoregressive decode ----------------
  init_tgt_kernel<<<(BSZ * NS + 255) / 256, 256, 0, s>>>(tgt, I(IN_BOS),
                                                         I(IN_PAD));
  for (int i = 0; i < NS; ++i) {
    int j = i + 1;
    long ctot = (long)BSZ * j * DMOD;
    content_kernel<<<(int)((ctot + 255) / 256), 256, 0, s>>>(
        tgt, F(IN_TEXTEMB), F(IN_POSQ), content, j);
    ln_kernel<<<BSZ * j, 128, 0, s>>>(content, F(IN_DEC_NCG), F(IN_DEC_NCB),
                                      cn, BSZ * j, DMOD);
    posq_bcast_kernel<<<(int)(((long)BSZ * DMOD + 255) / 256), 256, 0, s>>>(
        F(IN_POSQ), qb, 1, i);
    ln_kernel<<<BSZ, 128, 0, s>>>(qb, F(IN_DEC_NQG), F(IN_DEC_NQB), qn, BSZ,
                                  DMOD);
    // self attention (1 query x j content keys; mask is vacuous for AR step)
    gemm(s, qn, F(IN_DEC_SA_WQ), F(IN_DEC_SA_BQ), nullptr, qv, BSZ, DMOD, DMOD,
         DMOD, DMOD, DMOD);
    gemm(s, cn, F(IN_DEC_SA_WK), F(IN_DEC_SA_BK), nullptr, kb, BSZ * j, DMOD,
         DMOD, DMOD, DMOD, DMOD);
    gemm(s, cn, F(IN_DEC_SA_WV), F(IN_DEC_SA_BV), nullptr, vb, BSZ * j, DMOD,
         DMOD, DMOD, DMOD, DMOD);
    dec_attn_kernel<<<BSZ * DHEADS, 128, 0, s>>>(qv, kb, vb, att, BSZ, 1, j,
                                                 DMOD / DHEADS, DMOD, nullptr,
                                                 0);
    gemm(s, att, F(IN_DEC_SA_WO), F(IN_DEC_SA_BO), qb, xdec, BSZ, DMOD, DMOD,
         DMOD, DMOD, DMOD);
    // cross attention over memory
    ln_kernel<<<BSZ, 128, 0, s>>>(xdec, F(IN_DEC_N1G), F(IN_DEC_N1B), hdec,
                                  BSZ, DMOD);
    gemm(s, hdec, F(IN_DEC_CA_WQ), F(IN_DEC_CA_BQ), nullptr, qv, BSZ, DMOD,
         DMOD, DMOD, DMOD, DMOD);
    dec_attn_kernel<<<BSZ * DHEADS, 128, 0, s>>>(qv, mem_k, mem_v, att, BSZ, 1,
                                                 NTOK, DMOD / DHEADS, DMOD,
                                                 nullptr, 0);
    gemm(s, att, F(IN_DEC_CA_WO), F(IN_DEC_CA_BO), xdec, xdec, BSZ, DMOD, DMOD,
         DMOD, DMOD, DMOD);
    // MLP
    ln_kernel<<<BSZ, 128, 0, s>>>(xdec, F(IN_DEC_N2G), F(IN_DEC_N2B), hdec,
                                  BSZ, DMOD);
    gemm(s, hdec, F(IN_DEC_L1W), F(IN_DEC_L1B), nullptr, mlpd, BSZ, MLPD, DMOD,
         DMOD, MLPD, MLPD, 1, 0, 0, 0, 0, 0, 1);
    gemm(s, mlpd, F(IN_DEC_L2W), F(IN_DEC_L2B), xdec, xdec, BSZ, DMOD, MLPD,
         MLPD, DMOD, DMOD);
    ln_kernel<<<BSZ, 128, 0, s>>>(xdec, F(IN_DEC_NFG), F(IN_DEC_NFB), hdec,
                                  BSZ, DMOD);
    // head -> logits[:, i, :]
    gemm(s, hdec, F(IN_HEADW), F(IN_HEADB), nullptr, dout + (long)i * VOCAB,
         BSZ, VOCAB, DMOD, DMOD, VOCAB, NS * VOCAB);
    if (j < NS)
      argmax_step_kernel<<<1, 32, 0, s>>>(dout, tgt, i, j);
  }

  // ---------------- one refinement (cloze) pass ----------------
  retgt_kernel<<<1, 32, 0, s>>>(dout, tgt, I(IN_BOS));
  padmask_kernel<<<1, 32, 0, s>>>(tgt, I(IN_EOS), pmask);
  {
    const int M = BSZ * NS;
    long ctot = (long)M * DMOD;
    content_kernel<<<(int)((ctot + 255) / 256), 256, 0, s>>>(
        tgt, F(IN_TEXTEMB), F(IN_POSQ), content, NS);
    ln_kernel<<<M, 128, 0, s>>>(content, F(IN_DEC_NCG), F(IN_DEC_NCB), cn, M,
                                DMOD);
    posq_bcast_kernel<<<(int)((ctot + 255) / 256), 256, 0, s>>>(F(IN_POSQ), qb,
                                                                NS, 0);
    ln_kernel<<<M, 128, 0, s>>>(qb, F(IN_DEC_NQG), F(IN_DEC_NQB), qn, M, DMOD);
    gemm(s, qn, F(IN_DEC_SA_WQ), F(IN_DEC_SA_BQ), nullptr, qv, M, DMOD, DMOD,
         DMOD, DMOD, DMOD);
    gemm(s, cn, F(IN_DEC_SA_WK), F(IN_DEC_SA_BK), nullptr, kb, M, DMOD, DMOD,
         DMOD, DMOD, DMOD);
    gemm(s, cn, F(IN_DEC_SA_WV), F(IN_DEC_SA_BV), nullptr, vb, M, DMOD, DMOD,
         DMOD, DMOD, DMOD);
    dec_attn_kernel<<<BSZ * DHEADS * NS, 128, 0, s>>>(
        qv, kb, vb, att, BSZ, NS, NS, DMOD / DHEADS, DMOD, pmask, /*cloze*/ 1);
    gemm(s, att, F(IN_DEC_SA_WO), F(IN_DEC_SA_BO), qb, xdec, M, DMOD, DMOD,
         DMOD, DMOD, DMOD);
    ln_kernel<<<M, 128, 0, s>>>(xdec, F(IN_DEC_N1G), F(IN_DEC_N1B), hdec, M,
                                DMOD);
    gemm(s, hdec, F(IN_DEC_CA_WQ), F(IN_DEC_CA_BQ), nullptr, qv, M, DMOD, DMOD,
         DMOD, DMOD, DMOD);
    dec_attn_kernel<<<BSZ * DHEADS * NS, 128, 0, s>>>(
        qv, mem_k, mem_v, att, BSZ, NS, NTOK, DMOD / DHEADS, DMOD, nullptr, 0);
    gemm(s, att, F(IN_DEC_CA_WO), F(IN_DEC_CA_BO), xdec, xdec, M, DMOD, DMOD,
         DMOD, DMOD, DMOD);
    ln_kernel<<<M, 128, 0, s>>>(xdec, F(IN_DEC_N2G), F(IN_DEC_N2B), hdec, M,
                                DMOD);
    gemm(s, hdec, F(IN_DEC_L1W), F(IN_DEC_L1B), nullptr, mlpd, M, MLPD, DMOD,
         DMOD, MLPD, MLPD, 1, 0, 0, 0, 0, 0, 1);
    gemm(s, mlpd, F(IN_DEC_L2W), F(IN_DEC_L2B), xdec, xdec, M, DMOD, MLPD,
         MLPD, DMOD, DMOD);
    ln_kernel<<<M, 128, 0, s>>>(xdec, F(IN_DEC_NFG), F(IN_DEC_NFB), hdec, M,
                                DMOD);
    gemm(s, hdec, F(IN_HEADW), F(IN_HEADB), nullptr, dout, M, VOCAB, DMOD,
         DMOD, VOCAB, VOCAB);
  }
}
